// GraphDealModule_31447750542173
// MI455X (gfx1250) — matrix operand order repaired and checked by reading, not compiled, not run on hardware
//
#include <hip/hip_runtime.h>

// ---------------- problem constants ----------------
#define D512   512
#define N_EDGE 61440     // 256 graphs * 240 edges
#define N_NODE 4096      // 256 * 16
#define N_PRED 3840      // 256 * 15
#define EPG    240       // edges per graph
#define NPG    16        // nodes per graph

// ---------------- GEMM tiling ----------------
#define BM 128
#define BN 128
#define BK 32
#define LDA 40           // padded bf16 row stride in LDS (80B, 16B aligned)
#define LDB 40

typedef __bf16 bf16x16 __attribute__((ext_vector_type(16)));
typedef __bf16 bf16x2  __attribute__((ext_vector_type(2)));
typedef float  f32x8   __attribute__((ext_vector_type(8)));

union FragU {
  struct { uint4 lo, hi; } q;   // 32 bytes
  bf16x16 v;                    // 16 x bf16 = 32 bytes
};

__device__ __forceinline__ unsigned short f2bf(float f) {
  unsigned int u = __float_as_uint(f);
  u += 0x7FFFu + ((u >> 16) & 1u);      // round-to-nearest-even
  return (unsigned short)(u >> 16);
}

// pack two f32 into a dword of 2 bf16 (a -> low, b -> high)
__device__ __forceinline__ unsigned int pack2(float a, float b) {
#if __has_builtin(__builtin_amdgcn_cvt_pk_bf16_f32)
  bf16x2 h = __builtin_amdgcn_cvt_pk_bf16_f32(a, b);
  unsigned int u;
  __builtin_memcpy(&u, &h, 4);
  return u;
#elif __has_builtin(__builtin_amdgcn_perm)
  // round-to-nearest (ties away) then one v_perm_b32 grabs both high halves:
  // D = { (b+0x8000).hi16 , (a+0x8000).hi16 }
  const unsigned int ua = __float_as_uint(a) + 0x8000u;
  const unsigned int ub = __float_as_uint(b) + 0x8000u;
  return __builtin_amdgcn_perm(ub, ua, 0x07060302u);
#else
  return (unsigned int)f2bf(a) | ((unsigned int)f2bf(b) << 16);
#endif
}

// MODE 0: edge MLP rows = [visual[src[e]] | spatial[e] | visual[dst[e]]], K=1536
// MODE 1: node MLP rows = [visual[i]      | z_f[i]                    ], K=1024
// MODE 2: pred MLP rows = [nn[src[e]]     | spatial[e] | nn[dst[e]]   ], K=1536 (e = t_o_eid[row])
template <int MODE>
__global__ __launch_bounds__(256)
void gemm_bf16_wmma(const float* __restrict__ visual,
                    const float* __restrict__ spatial,
                    const float* __restrict__ W,      // [K, 512] row-major
                    const float* __restrict__ bias,   // [512]
                    const float* __restrict__ extra,  // z_f (MODE1) or new_n_f (MODE2)
                    const int*   __restrict__ srcIdx,
                    const int*   __restrict__ dstIdx,
                    const int*   __restrict__ toe,
                    float*       __restrict__ out,    // [M, 512] f32 (post bias+ReLU)
                    int K)
{
  __shared__ unsigned short As[BM * LDA];
  __shared__ unsigned short Bs[BN * LDB];

  const int tid  = threadIdx.x;
  const int lane = tid & 31;
  const int wv   = tid >> 5;      // 8 waves
  const int wm   = wv & 3;        // 4 M-wave groups (32 rows each)
  const int wn   = wv >> 2;       // 2 N-wave groups (64 cols each)
  const int hb   = lane >> 4;     // half-wave select
  const int l15  = lane & 15;

  const int blockM = blockIdx.y * BM;
  const int blockN = blockIdx.x * BN;

  // ---- per-thread gathered A row base pointers (loader role) ----
  const int arow  = blockM + (tid >> 1);
  const int ahalf = tid & 1;               // which 16-wide half of the 32-wide k chunk
  const float* pA0; const float* pA1; const float* pA2;
  if constexpr (MODE == 0) {
    pA0 = visual  + (size_t)srcIdx[arow] * D512;
    pA1 = spatial + (size_t)arow        * D512;
    pA2 = visual  + (size_t)dstIdx[arow] * D512;
  } else if constexpr (MODE == 1) {
    pA0 = visual + (size_t)arow * D512;
    pA1 = extra  + (size_t)arow * D512;
    pA2 = pA1;
  } else {
    const int e = toe[arow];
    pA0 = extra   + (size_t)srcIdx[e] * D512;
    pA1 = spatial + (size_t)e         * D512;
    pA2 = extra   + (size_t)dstIdx[e] * D512;
  }

  // B loader role: thread handles k-pair (bkq, bkq+1) x 8 n columns
  const int bkq  = (tid & 15) * 2;   // even k
  const int bgrp = tid >> 4;         // 0..15 -> n range [bgrp*8, bgrp*8+8)

  auto loadA = [&](int kBase, float4 (&v)[4]) {
    const int seg = kBase >> 9;                       // 512-wide segments
    const float* base = (seg == 0) ? pA0 : (seg == 1) ? pA1 : pA2;
    const float* p = base + (kBase & 511) + ahalf * 16;
    v[0] = ((const float4*)p)[0];
    v[1] = ((const float4*)p)[1];
    v[2] = ((const float4*)p)[2];
    v[3] = ((const float4*)p)[3];
  };
  auto loadB = [&](int kBase, float4 (&v)[4]) {
    const float* p0 = W + (size_t)(kBase + bkq) * D512 + blockN + bgrp * 8;
    v[0] = ((const float4*)p0)[0];
    v[1] = ((const float4*)p0)[1];
    v[2] = ((const float4*)(p0 + D512))[0];
    v[3] = ((const float4*)(p0 + D512))[1];
  };
  auto storeA = [&](const float4 (&v)[4]) {
    uint4* q = (uint4*)&As[(tid >> 1) * LDA + ahalf * 16];
    q[0] = make_uint4(pack2(v[0].x, v[0].y), pack2(v[0].z, v[0].w),
                      pack2(v[1].x, v[1].y), pack2(v[1].z, v[1].w));
    q[1] = make_uint4(pack2(v[2].x, v[2].y), pack2(v[2].z, v[2].w),
                      pack2(v[3].x, v[3].y), pack2(v[3].z, v[3].w));
  };
  auto storeB = [&](const float4 (&v)[4]) {
    const float r0[8] = { v[0].x, v[0].y, v[0].z, v[0].w,
                          v[1].x, v[1].y, v[1].z, v[1].w };
    const float r1[8] = { v[2].x, v[2].y, v[2].z, v[2].w,
                          v[3].x, v[3].y, v[3].z, v[3].w };
    #pragma unroll
    for (int i = 0; i < 8; ++i)   // packed (k, k+1) pair per b32 store
      *(unsigned int*)&Bs[(bgrp * 8 + i) * LDB + bkq] = pack2(r0[i], r1[i]);
  };

  f32x8 acc[2][4] = {};

  // ---- software pipeline: global loads of tile k+1 overlap WMMA of tile k ----
  float4 aReg[4], bReg[4];
  loadA(0, aReg);
  loadB(0, bReg);

  for (int kBase = 0; kBase < K; kBase += BK) {
    storeA(aReg);
    storeB(bReg);
    __syncthreads();

    const int kNext = kBase + BK;
    if (kNext < K) {          // issue next tile's global loads now (in flight
      loadA(kNext, aReg);     // while the WMMA chain below executes)
      loadB(kNext, bReg);
    }

    // ---- fragment loads (contiguous k order: lane half hb holds K = hb*16 .. +16 for both A and B) ----
    bf16x16 afrag[2];
    bf16x16 bfrag[4];
    #pragma unroll
    for (int mi = 0; mi < 2; ++mi) {
      const unsigned short* ap = &As[(wm * 32 + mi * 16 + l15) * LDA];
      FragU u;
      u.q.lo = *(const uint4*)(ap + hb * 16);       // K = hb*16 .. +8
      u.q.hi = *(const uint4*)(ap + hb * 16 + 8);   // K = hb*16+8 .. +8
      afrag[mi] = u.v;
    }
    #pragma unroll
    for (int ni = 0; ni < 4; ++ni) {
      const unsigned short* bp = &Bs[(wn * 64 + ni * 16 + l15) * LDB + hb * 16];
      FragU u;
      u.q.lo = *(const uint4*)(bp);                 // K = hb*16 .. +8
      u.q.hi = *(const uint4*)(bp + 8);             // K = hb*16+8 .. +8
      bfrag[ni] = u.v;
    }
    #pragma unroll
    for (int mi = 0; mi < 2; ++mi)
      #pragma unroll
      for (int ni = 0; ni < 4; ++ni)
        acc[mi][ni] = __builtin_amdgcn_wmma_f32_16x16x32_bf16(
            false, afrag[mi], false, bfrag[ni], (short)0, acc[mi][ni],
            false, false);
    __syncthreads();
  }

  // ---- epilogue: bias + ReLU, f32 stores ----
  #pragma unroll
  for (int mi = 0; mi < 2; ++mi) {
    const int rbase = blockM + wm * 32 + mi * 16 + hb * 8;
    #pragma unroll
    for (int ni = 0; ni < 4; ++ni) {
      const int n = blockN + wn * 64 + ni * 16 + l15;
      const float bv = bias[n];
      #pragma unroll
      for (int i = 0; i < 8; ++i) {
        float v = acc[mi][ni][i] + bv;
        v = v > 0.f ? v : 0.f;
        out[(size_t)(rbase + i) * D512 + n] = v;
      }
    }
  }
}

// ---------------- per-graph attention + segment softmax + aggregation ----------------
// One block per graph (256 threads, 8 waves). Edge j of a graph: src s = j/15,
// dst d = (j%15) + (j%15 >= s). Each dst node has exactly 15 in-edges.
__global__ __launch_bounds__(256)
void attn_softmax_agg(const float* __restrict__ ef,      // [E,512] relu'd edge feats
                      const float* __restrict__ visual,  // [N,512]
                      const float* __restrict__ Wattn,   // [512]
                      const float* __restrict__ battn,   // [1]
                      float*       __restrict__ zf)      // [N,512] out
{
  __shared__ float a_sh[EPG];
  __shared__ float alpha_sh[EPG];
  const int g = blockIdx.x;
  const int lane = threadIdx.x & 31;
  const int w = threadIdx.x >> 5;

  // phase 1: attention logits a[j] = relu(e_f . W_attn + b)
  for (int jj = 0; jj < EPG / 8; ++jj) {
    const int j = w * (EPG / 8) + jj;
    const size_t e = (size_t)(g * EPG + j) * D512;
    float s = 0.f;
    #pragma unroll
    for (int c = 0; c < 16; ++c) {
      const int dim = c * 32 + lane;
      s += ef[e + dim] * Wattn[dim];
    }
    #pragma unroll
    for (int off = 16; off > 0; off >>= 1) s += __shfl_xor(s, off, 32);
    if (lane == 0) {
      float v = s + battn[0];
      a_sh[j] = v > 0.f ? v : 0.f;
    }
  }
  __syncthreads();

  // phase 2: softmax over the 15 in-edges of each of 16 dst nodes
  if (threadIdx.x < NPG) {
    const int dn = threadIdx.x;
    float m = -1e30f;
    for (int s = 0; s < NPG; ++s) {
      if (s == dn) continue;
      const int j = s * 15 + dn - (dn > s ? 1 : 0);
      m = fmaxf(m, a_sh[j]);
    }
    float den = 0.f;
    for (int s = 0; s < NPG; ++s) {
      if (s == dn) continue;
      const int j = s * 15 + dn - (dn > s ? 1 : 0);
      den += __expf(a_sh[j] - m);
    }
    const float inv = 1.0f / den;
    for (int s = 0; s < NPG; ++s) {
      if (s == dn) continue;
      const int j = s * 15 + dn - (dn > s ? 1 : 0);
      alpha_sh[j] = __expf(a_sh[j] - m) * inv;
    }
  }
  __syncthreads();

  // phase 3: z_f[d] = sum_s alpha * (visual[src] + e_f[edge])
  for (int t = 0; t < 2; ++t) {
    const int dn = w * 2 + t;
    for (int c = 0; c < 16; ++c) {
      const int dim = c * 32 + lane;
      float acc = 0.f;
      for (int s = 0; s < NPG; ++s) {
        if (s == dn) continue;
        const int j = s * 15 + dn - (dn > s ? 1 : 0);
        const size_t e = (size_t)(g * EPG + j);
        acc += alpha_sh[j] * (visual[(size_t)(g * NPG + s) * D512 + dim] +
                              ef[e * D512 + dim]);
      }
      zf[(size_t)(g * NPG + dn) * D512 + dim] = acc;
    }
  }
}

extern "C" void kernel_launch(void* const* d_in, const int* in_sizes, int n_in,
                              void* d_out, int out_size, void* d_ws, size_t ws_size,
                              hipStream_t stream) {
  (void)in_sizes; (void)n_in; (void)out_size; (void)ws_size;
  const float* visual  = (const float*)d_in[0];
  const float* spatial = (const float*)d_in[1];
  const float* W_edge  = (const float*)d_in[2];
  const float* b_edge  = (const float*)d_in[3];
  const float* W_attn  = (const float*)d_in[4];
  const float* b_attn  = (const float*)d_in[5];
  const float* W_node  = (const float*)d_in[6];
  const float* b_node  = (const float*)d_in[7];
  const float* W_pred  = (const float*)d_in[8];
  const float* b_pred  = (const float*)d_in[9];
  const int*   src     = (const int*)d_in[10];
  const int*   dst     = (const int*)d_in[11];
  const int*   toe     = (const int*)d_in[12];
  float* out = (float*)d_out;

  // workspace: e_f (f32 [E,512]) | z_f (f32 [N,512]) | new_n_f (f32 [N,512])
  float* ef = (float*)d_ws;
  float* zf = ef + (size_t)N_EDGE * D512;
  float* nn = zf + (size_t)N_NODE * D512;

  // 1) edge MLP (gather-fused GEMM, M=61440, K=1536, N=512)
  gemm_bf16_wmma<0><<<dim3(D512 / BN, N_EDGE / BM), 256, 0, stream>>>(
      visual, spatial, W_edge, b_edge, nullptr, src, dst, nullptr, ef, 3 * D512);

  // 2) attention logit + per-dst softmax + weighted aggregation (one block/graph)
  attn_softmax_agg<<<N_NODE / NPG, 256, 0, stream>>>(ef, visual, W_attn, b_attn, zf);

  // 3) node MLP (M=4096, K=1024, N=512)
  gemm_bf16_wmma<1><<<dim3(D512 / BN, N_NODE / BM), 256, 0, stream>>>(
      visual, spatial, W_node, b_node, zf, src, dst, nullptr, nn, 2 * D512);

  // 4) interaction predictor on target->other edges (M=3840, K=1536, N=512)
  gemm_bf16_wmma<2><<<dim3(D512 / BN, N_PRED / BM), 256, 0, stream>>>(
      visual, spatial, W_pred, b_pred, nn, src, dst, toe, out, 3 * D512);
}
